// NNG_56942676411057
// MI455X (gfx1250) — matrix operand order repaired and checked by reading, not compiled, not run on hardware
//
#include <hip/hip_runtime.h>
#include <hip/hip_bf16.h>

// ---------------------------------------------------------------------------
// GCN 2-layer forward on gfx1250 (MI455X).
//   support1 = X @ W1                 (bf16 WMMA, f32 accum)
//   h        = scatter_add(support1)  (wave-per-edge f32 atomics, L2-resident)
//   support2 = relu(h) @ W2           (bf16 WMMA, relu fused into A staging)
//   out      = scatter_add(support2)
// ---------------------------------------------------------------------------

typedef __attribute__((ext_vector_type(16))) __bf16 v16bf;
typedef __attribute__((ext_vector_type(8)))  float  v8f;

union FragAB { unsigned int u[8]; v16bf v; };

__device__ __forceinline__ unsigned int bf16_rne(float f) {
    unsigned int u = __float_as_uint(f);
    return ((u + 0x7FFFu + ((u >> 16) & 1u)) >> 16) & 0xFFFFu;
}
__device__ __forceinline__ unsigned int pack_bf16(float lo, float hi) {
    return bf16_rne(lo) | (bf16_rne(hi) << 16);
}

// --- tiny f32 -> bf16 weight conversion ------------------------------------
__global__ void cvt_f32_to_bf16(const float* __restrict__ src,
                                unsigned short* __restrict__ dst, int n) {
    int i = blockIdx.x * blockDim.x + threadIdx.x;
    if (i < n) dst[i] = (unsigned short)bf16_rne(src[i]);
}

// --- zero fill --------------------------------------------------------------
__global__ void zero_f32(float* __restrict__ p, size_t n) {
    size_t i = (size_t)blockIdx.x * blockDim.x + threadIdx.x;
    size_t stride = (size_t)gridDim.x * blockDim.x;
    for (; i < n; i += stride) p[i] = 0.0f;
}

// --- dense transform: out[M x ldN] = (relu?)X[M x K] @ Wbf[K x ldN] ---------
// Block = 128 threads = 4 waves. Block covers 16 rows x 64 cols; wave w owns
// the 16x16 tile at colBase + 16*w. Full-K staged in LDS as packed bf16 pairs
// so every fragment dword is a single ds_load_b32.
template <int K, bool RELU>
__global__ __launch_bounds__(128) void gcn_gemm_wmma(
    const float* __restrict__ X, const unsigned short* __restrict__ Wbf,
    float* __restrict__ out, int ldN) {
    __shared__ unsigned int lds_x[(K / 2) * 16];   // [kpair][m]
    __shared__ unsigned int lds_w[(K / 2) * 64];   // [kpair][n]

    const int tid     = threadIdx.x;
    const int rowBase = blockIdx.x * 16;
    const int colBase = blockIdx.y * 64;

    // Stage X tile (16 x K) -> packed bf16 K-pairs. Consecutive threads read
    // consecutive K (contiguous 8B global loads).
    for (int idx = tid; idx < 16 * (K / 2); idx += 128) {
        int m  = idx / (K / 2);
        int kp = idx % (K / 2);
        const float* p = X + (size_t)(rowBase + m) * K + 2 * kp;
        float f0 = p[0], f1 = p[1];
        if (RELU) { f0 = fmaxf(f0, 0.0f); f1 = fmaxf(f1, 0.0f); }
        lds_x[kp * 16 + m] = pack_bf16(f0, f1);
    }
    // Stage W tile (K x 64) -> pairs along K: lds_w[kp*64+n] = {W[2kp][c], W[2kp+1][c]}
    for (int idx = tid; idx < (K / 2) * 64; idx += 128) {
        int kp = idx / 64;
        int n  = idx % 64;
        int c  = colBase + n;
        unsigned int lo = Wbf[(size_t)(2 * kp) * ldN + c];
        unsigned int hi = Wbf[(size_t)(2 * kp + 1) * ldN + c];
        lds_w[idx] = lo | (hi << 16);
    }
    __syncthreads();

    const int wave = tid >> 5;
    const int lane = tid & 31;
    const int m    = lane & 15;      // row (A) / col (B,D) within tile
    const int hi   = lane >> 4;      // lane half selects K sub-block

    FragAB a, b;
    v8f c = {};
    for (int k0 = 0; k0 < K; k0 += 32) {
        const int kb = k0 >> 1;  // K-pair base
        // A 16x32 bf16 layout: VGPR j: K = k0 + hi*8 + 2j ; VGPR 4+j: +16
#pragma unroll
        for (int j = 0; j < 4; ++j) {
            a.u[j]     = lds_x[(kb + hi * 4 + j) * 16 + m];
            a.u[4 + j] = lds_x[(kb + 8 + hi * 4 + j) * 16 + m];
        }
        // B 32x16 bf16 layout: VGPR r: K = k0 + (r>>2)*16 + hi*8 + 2(r&3), N = lane&15
#pragma unroll
        for (int r = 0; r < 8; ++r)
            b.u[r] = lds_w[(kb + (r >> 2) * 8 + hi * 4 + (r & 3)) * 64 + wave * 16 + m];

        c = __builtin_amdgcn_wmma_f32_16x16x32_bf16(
            false, a.v, false, b.v, (short)0, c, false, false);
    }

    // D layout: VGPR r -> row (r + 8*hi), col = lane&15
    float* o = out + (size_t)(rowBase + hi * 8) * ldN + colBase + wave * 16 + m;
#pragma unroll
    for (int r = 0; r < 8; ++r) o[(size_t)r * ldN] = c[r];
}

// --- edge scatter: dst[row[e]][:] += src[col[e]][:] -------------------------
// One wave per edge: indices are wave-uniform (one broadcast load), payload is
// a coalesced b128/b64 gather + F/32 scalar f32 atomics per lane. Working set
// (<130 MB) is resident in the 192 MB L2, so this is L2-atomic bound.
template <int F>
__global__ __launch_bounds__(256) void scatter_add_edges(
    const float* __restrict__ src, const int* __restrict__ row,
    const int* __restrict__ col, float* __restrict__ dst, int nEdges) {
    constexpr int V = F / 32;  // floats per lane: 4 (F=128) or 2 (F=64)
    const int wavesPerBlock = blockDim.x >> 5;
    const int lane   = threadIdx.x & 31;
    int e            = blockIdx.x * wavesPerBlock + (threadIdx.x >> 5);
    const int stride = gridDim.x * wavesPerBlock;

    for (; e < nEdges; e += stride) {
        const int r = row[e];
        const int c = col[e];
        const float* s = src + (size_t)c * F + lane * V;
        float* d       = dst + (size_t)r * F + lane * V;
        if constexpr (V == 4) {
            float4 v = *(const float4*)s;
            atomicAdd(d + 0, v.x);
            atomicAdd(d + 1, v.y);
            atomicAdd(d + 2, v.z);
            atomicAdd(d + 3, v.w);
        } else {
            float2 v = *(const float2*)s;
            atomicAdd(d + 0, v.x);
            atomicAdd(d + 1, v.y);
        }
    }
}

extern "C" void kernel_launch(void* const* d_in, const int* in_sizes, int n_in,
                              void* d_out, int out_size, void* d_ws, size_t ws_size,
                              hipStream_t stream) {
    constexpr int IN_DIM = 256, HIDDEN = 128, OUT_DIM = 64;

    const float* features = (const float*)d_in[0];   // [M, 256] f32
    const int*   edge_idx = (const int*)d_in[1];     // [2, E]   (harness int)
    const float* W1       = (const float*)d_in[2];   // [256, 128] f32
    const float* W2       = (const float*)d_in[3];   // [128, 64]  f32
    float*       out      = (float*)d_out;           // [M, 64]  f32

    const int M  = in_sizes[0] / IN_DIM;             // 100000 (divisible by 16)
    const int nE = in_sizes[1] / 2;                  // 3.2M
    const int* rowI = edge_idx;
    const int* colI = edge_idx + nE;

    // Workspace layout (peak ~102.5 MB):
    float* support1 = (float*)d_ws;                          // M*128 f32
    float* h        = support1 + (size_t)M * HIDDEN;         // M*128 f32
    float* support2 = (float*)d_ws;                          // reuse: M*64 f32
    unsigned short* w1bf = (unsigned short*)(h + (size_t)M * HIDDEN);
    unsigned short* w2bf = w1bf + IN_DIM * HIDDEN;

    // 1) bf16 weights
    cvt_f32_to_bf16<<<(IN_DIM * HIDDEN + 255) / 256, 256, 0, stream>>>(
        W1, w1bf, IN_DIM * HIDDEN);
    cvt_f32_to_bf16<<<(HIDDEN * OUT_DIM + 255) / 256, 256, 0, stream>>>(
        W2, w2bf, HIDDEN * OUT_DIM);

    // 2) support1 = X @ W1   (grid: 6250 x 2, each block 16 rows x 64 cols)
    gcn_gemm_wmma<IN_DIM, false><<<dim3(M / 16, HIDDEN / 64), 128, 0, stream>>>(
        features, w1bf, support1, HIDDEN);

    // 3) h = scatter_add(support1)
    zero_f32<<<2048, 256, 0, stream>>>(h, (size_t)M * HIDDEN);
    scatter_add_edges<HIDDEN><<<4096, 256, 0, stream>>>(support1, rowI, colI, h, nE);

    // 4) support2 = relu(h) @ W2   (relu fused into A-tile staging)
    gcn_gemm_wmma<HIDDEN, true><<<dim3(M / 16, OUT_DIM / 64), 128, 0, stream>>>(
        h, w2bf, support2, OUT_DIM);

    // 5) out = scatter_add(support2)
    zero_f32<<<2048, 256, 0, stream>>>(out, (size_t)M * OUT_DIM);
    scatter_add_edges<OUT_DIM><<<4096, 256, 0, stream>>>(support2, rowI, colI, out, nE);
}
